// ActorCriticGAT_74844100100550
// MI455X (gfx1250) — matrix operand order repaired and checked by reading, not compiled, not run on hardware
//
#include <hip/hip_runtime.h>
#include <hip/hip_bf16.h>

typedef __attribute__((ext_vector_type(16))) _Float16 v16h;
typedef __attribute__((ext_vector_type(8)))  float    v8f;

#define HID   64
#define HEADS 4

// ---------- helpers: order-preserving float<->uint for atomicMax ----------
__device__ __forceinline__ unsigned fenc(float f) {
    unsigned u = __float_as_uint(f);
    return (u & 0x80000000u) ? ~u : (u | 0x80000000u);
}
__device__ __forceinline__ float fdec(unsigned e) {
    unsigned u = (e & 0x80000000u) ? (e ^ 0x80000000u) : ~e;
    return __uint_as_float(u);
}

// ---------- init ----------
__global__ void k_zero(unsigned* p, long n) {
    long t = (long)blockIdx.x * blockDim.x + threadIdx.x;
    if (t < n) p[t] = 0u;
}

// ---------- layer 1: h1 = x @ W1  (K=3, scalar; memory bound) ----------
__global__ void k_gemm1(const float* __restrict__ x, const float* __restrict__ W1,
                        float* __restrict__ h1, int N) {
    int t = blockIdx.x * blockDim.x + threadIdx.x;
    if (t >= N * 256) return;
    int n = t >> 8, c = t & 255;
    float v = x[n * 3 + 0] * W1[0 * 256 + c]
            + x[n * 3 + 1] * W1[1 * 256 + c]
            + x[n * 3 + 2] * W1[2 * 256 + c];
    h1[t] = v;
}

// ---------- per-node attention coefficients, layer 1 ----------
__global__ void k_alpha1(const float* __restrict__ h1, const float* __restrict__ a_s,
                         const float* __restrict__ a_d, float* __restrict__ as_o,
                         float* __restrict__ ad_o, int N) {
    int t = blockIdx.x * blockDim.x + threadIdx.x;
    if (t >= N * HEADS) return;
    int n = t >> 2, h = t & 3;
    const float* hp = h1 + n * 256 + h * HID;
    float s = 0.f, d = 0.f;
    #pragma unroll 8
    for (int c = 0; c < HID; ++c) { s += hp[c] * a_s[h * HID + c]; d += hp[c] * a_d[h * HID + c]; }
    as_o[t] = s; ad_o[t] = d;
}

// ---------- edge pass A: e = leaky_relu(as[src]+ad[dst]); segment max ----------
__global__ void k_edge_max1(const int* __restrict__ src, const int* __restrict__ dst,
                            const float* __restrict__ as_, const float* __restrict__ ad_,
                            float* __restrict__ e_out, unsigned* __restrict__ m, int E) {
    int t = blockIdx.x * blockDim.x + threadIdx.x;
    if (t >= E * HEADS) return;
    int e = t >> 2, h = t & 3;
    float v = as_[src[e] * HEADS + h] + ad_[dst[e] * HEADS + h];
    v = v > 0.f ? v : 0.2f * v;
    e_out[t] = v;
    atomicMax(&m[dst[e] * HEADS + h], fenc(v));
}

// ---------- edge pass B: z[dst] += exp(e - m[dst]) ----------
__global__ void k_edge_expsum1(const int* __restrict__ dst, const float* __restrict__ e_in,
                               const unsigned* __restrict__ m, float* __restrict__ z, int E) {
    int t = blockIdx.x * blockDim.x + threadIdx.x;
    if (t >= E * HEADS) return;
    int e = t >> 2, h = t & 3;
    float p = expf(e_in[t] - fdec(m[dst[e] * HEADS + h]));
    atomicAdd(&z[dst[e] * HEADS + h], p);
}

// ---------- edge pass C: acc[dst] += alpha * h1[src]  (4 ch per thread) ----------
__global__ void k_edge_msg1(const int* __restrict__ src, const int* __restrict__ dst,
                            const float* __restrict__ e_in, const unsigned* __restrict__ m,
                            const float* __restrict__ z, const float* __restrict__ h1,
                            float* __restrict__ acc, int E) {
    long t = (long)blockIdx.x * blockDim.x + threadIdx.x;
    if (t >= (long)E * HEADS * 16) return;
    int g  = (int)(t & 15);
    int eh = (int)(t >> 4);
    int h = eh & 3, e = eh >> 2;
    int s = src[e], d = dst[e];
    float alpha = expf(e_in[eh] - fdec(m[d * HEADS + h])) / (z[d * HEADS + h] + 1e-16f);
    const float4 hv = *(const float4*)(h1 + s * 256 + h * HID + g * 4);
    float* o = acc + d * 256 + h * HID + g * 4;
    atomicAdd(o + 0, hv.x * alpha);
    atomicAdd(o + 1, hv.y * alpha);
    atomicAdd(o + 2, hv.z * alpha);
    atomicAdd(o + 3, hv.w * alpha);
}

// ---------- x = relu(acc + b1)   (in place on acc) ----------
__global__ void k_finx(float* __restrict__ acc, const float* __restrict__ b1, int N) {
    int t = blockIdx.x * blockDim.x + threadIdx.x;
    if (t >= N * 256) return;
    float v = acc[t] + b1[t & 255];
    acc[t] = v > 0.f ? v : 0.f;
}

// ---------- pack W2 [256,64] f32 -> f16 in WMMA B-register order ----------
// element t -> j=t&15 (K within block of 16), lane=(t>>4)&31, nt=(t>>9)&3, chunk=t>>11
// lane: N = nt*16 + (lane&15), K-block = 16*(lane>>4); K = chunk*32 + 16*(lane>>4) + j
__global__ void k_packW2(const float* __restrict__ W2, _Float16* __restrict__ W2p) {
    int t = blockIdx.x * blockDim.x + threadIdx.x;
    if (t >= 8 * 4 * 32 * 16) return;
    int j     = t & 15;
    int lane  = (t >> 4) & 31;
    int nt    = (t >> 9) & 3;
    int chunk = t >> 11;
    int ncol  = nt * 16 + (lane & 15);
    int kk    = chunk * 32 + 16 * (lane >> 4) + j;
    W2p[t] = (_Float16)W2[kk * HID + ncol];
}

// ---------- layer 2 GEMM via WMMA: h2 = X @ W2, X:[N,256], W2 packed f16 ----------
// One wave computes a full M=16 x N=64 output stripe (4 accumulators, A reused 4x).
__global__ void k_gemm2_wmma(const float* __restrict__ X, const _Float16* __restrict__ W2p,
                             float* __restrict__ H2, int N) {
    int wave = blockIdx.x * (blockDim.x >> 5) + (threadIdx.x >> 5);
    int lane = threadIdx.x & 31;
    int mtiles = (N + 15) >> 4;
    if (wave >= mtiles) return;

    int half = lane >> 4;        // K-block selector within layouts
    int l16  = lane & 15;
    int mrow = wave * 16 + l16;
    if (mrow >= N) mrow = N - 1;           // clamp: surplus rows are never stored
    const float* Xrow = X + (long)mrow * 256;

    v8f c0 = {}, c1 = {}, c2 = {}, c3 = {};
    for (int k0 = 0; k0 < 256; k0 += 32) {
        // ---- A tile: per lane, two contiguous runs of 8 floats ----
        // a[i]   <- X[mrow][k0 + 16*half + i]       (i=0..7)
        // a[8+i] <- X[mrow][k0 + 16*half + 8 + i]   (i=0..7)
        const float4* p0 = (const float4*)(Xrow + k0 + 16 * half);
        const float4* p1 = (const float4*)(Xrow + k0 + 16 * half + 8);
        float4 f0 = p0[0], f1 = p0[1], f2 = p1[0], f3 = p1[1];
        v16h a;
        a[0]  = (_Float16)f0.x; a[1]  = (_Float16)f0.y; a[2]  = (_Float16)f0.z; a[3]  = (_Float16)f0.w;
        a[4]  = (_Float16)f1.x; a[5]  = (_Float16)f1.y; a[6]  = (_Float16)f1.z; a[7]  = (_Float16)f1.w;
        a[8]  = (_Float16)f2.x; a[9]  = (_Float16)f2.y; a[10] = (_Float16)f2.z; a[11] = (_Float16)f2.w;
        a[12] = (_Float16)f3.x; a[13] = (_Float16)f3.y; a[14] = (_Float16)f3.z; a[15] = (_Float16)f3.w;

        // ---- B tiles: packed, 32 contiguous bytes per lane per n-tile ----
        const v16h* bp = (const v16h*)W2p + (k0 >> 5) * 128 + lane;
        v16h b0 = bp[0];
        v16h b1 = bp[32];
        v16h b2 = bp[64];
        v16h b3 = bp[96];

        c0 = __builtin_amdgcn_wmma_f32_16x16x32_f16(false, a, false, b0, (short)0, c0, false, false);
        c1 = __builtin_amdgcn_wmma_f32_16x16x32_f16(false, a, false, b1, (short)0, c1, false, false);
        c2 = __builtin_amdgcn_wmma_f32_16x16x32_f16(false, a, false, b2, (short)0, c2, false, false);
        c3 = __builtin_amdgcn_wmma_f32_16x16x32_f16(false, a, false, b3, (short)0, c3, false, false);
    }

    // D layout: VGPR r -> M = r + 8*half, N = l16 (+16*nt)
    #pragma unroll
    for (int r = 0; r < 8; ++r) {
        int mm = wave * 16 + r + 8 * half;
        if (mm < N) {
            float* orow = H2 + (long)mm * HID + l16;
            orow[0]  = c0[r];
            orow[16] = c1[r];
            orow[32] = c2[r];
            orow[48] = c3[r];
        }
    }
}

// ---------- per-node attention coefficients, layer 2 (H=1) ----------
__global__ void k_alpha2(const float* __restrict__ h2, const float* __restrict__ a_s,
                         const float* __restrict__ a_d, float* __restrict__ as_o,
                         float* __restrict__ ad_o, int N) {
    int n = blockIdx.x * blockDim.x + threadIdx.x;
    if (n >= N) return;
    float s = 0.f, d = 0.f;
    #pragma unroll 8
    for (int c = 0; c < HID; ++c) { s += h2[n * HID + c] * a_s[c]; d += h2[n * HID + c] * a_d[c]; }
    as_o[n] = s; ad_o[n] = d;
}

__global__ void k_edge_max2(const int* __restrict__ src, const int* __restrict__ dst,
                            const float* __restrict__ as_, const float* __restrict__ ad_,
                            float* __restrict__ e_out, unsigned* __restrict__ m, int E) {
    int e = blockIdx.x * blockDim.x + threadIdx.x;
    if (e >= E) return;
    float v = as_[src[e]] + ad_[dst[e]];
    v = v > 0.f ? v : 0.2f * v;
    e_out[e] = v;
    atomicMax(&m[dst[e]], fenc(v));
}

__global__ void k_edge_expsum2(const int* __restrict__ dst, const float* __restrict__ e_in,
                               const unsigned* __restrict__ m, float* __restrict__ z, int E) {
    int e = blockIdx.x * blockDim.x + threadIdx.x;
    if (e >= E) return;
    float p = expf(e_in[e] - fdec(m[dst[e]]));
    atomicAdd(&z[dst[e]], p);
}

__global__ void k_edge_msg2(const int* __restrict__ src, const int* __restrict__ dst,
                            const float* __restrict__ e_in, const unsigned* __restrict__ m,
                            const float* __restrict__ z, const float* __restrict__ h2,
                            float* __restrict__ acc, int E) {
    long t = (long)blockIdx.x * blockDim.x + threadIdx.x;
    if (t >= (long)E * 16) return;
    int g = (int)(t & 15);
    int e = (int)(t >> 4);
    int s = src[e], d = dst[e];
    float alpha = expf(e_in[e] - fdec(m[d])) / (z[d] + 1e-16f);
    const float4 hv = *(const float4*)(h2 + s * HID + g * 4);
    float* o = acc + d * HID + g * 4;
    atomicAdd(o + 0, hv.x * alpha);
    atomicAdd(o + 1, hv.y * alpha);
    atomicAdd(o + 2, hv.z * alpha);
    atomicAdd(o + 3, hv.w * alpha);
}

// ---------- emb = acc + b2 ----------
__global__ void k_finemb(float* __restrict__ acc, const float* __restrict__ b2, int N) {
    int t = blockIdx.x * blockDim.x + threadIdx.x;
    if (t >= N * HID) return;
    acc[t] = acc[t] + b2[t & (HID - 1)];
}

// ---------- column sum for graph embedding ----------
__global__ void k_colsum(const float* __restrict__ emb, float* __restrict__ gsum, int N) {
    int c = threadIdx.x;                 // 64 threads
    float s = 0.f;
    for (int n = blockIdx.x; n < N; n += gridDim.x) s += emb[n * HID + c];
    atomicAdd(&gsum[c], s);
}

// ---------- heads: value, policy, softmax ----------
__global__ void k_heads(const float* __restrict__ gsum, const float* __restrict__ emb,
                        const float* __restrict__ Wv, const float* __restrict__ bv,
                        const float* __restrict__ Wp, const float* __restrict__ bp,
                        const int* __restrict__ cur, float* __restrict__ out, int N) {
    __shared__ float ge[HID];
    int c = threadIdx.x;
    if (c < HID) ge[c] = gsum[c] / (float)N;
    __syncthreads();
    if (c == 0) {
        float sv = bv[0];
        for (int i = 0; i < HID; ++i) sv += ge[i] * Wv[i];
        int ci = cur[0];
        float l0 = bp[0], l1 = bp[1];
        for (int i = 0; i < HID; ++i) {
            float ev = emb[ci * HID + i];
            l0 += ev * Wp[i * 2 + 0];
            l1 += ev * Wp[i * 2 + 1];
        }
        float mx = fmaxf(l0, l1);
        float p0 = expf(l0 - mx), p1 = expf(l1 - mx);
        float inv = 1.f / (p0 + p1);
        out[0] = p0 * inv;
        out[1] = p1 * inv;
        out[2] = sv;
    }
}

extern "C" void kernel_launch(void* const* d_in, const int* in_sizes, int n_in,
                              void* d_out, int out_size, void* d_ws, size_t ws_size,
                              hipStream_t stream) {
    const float* x_in = (const float*)d_in[0];
    const float* W1   = (const float*)d_in[1];
    const float* aS1  = (const float*)d_in[2];
    const float* aD1  = (const float*)d_in[3];
    const float* b1   = (const float*)d_in[4];
    const float* W2   = (const float*)d_in[5];
    const float* aS2  = (const float*)d_in[6];
    const float* aD2  = (const float*)d_in[7];
    const float* b2   = (const float*)d_in[8];
    const float* Wp   = (const float*)d_in[9];
    const float* bp   = (const float*)d_in[10];
    const float* Wv   = (const float*)d_in[11];
    const float* bv   = (const float*)d_in[12];
    const int*   eidx = (const int*)d_in[13];
    const int*   cur  = (const int*)d_in[14];
    float* out = (float*)d_out;

    const int N = in_sizes[0] / 3;
    const int E = in_sizes[13] / 2;
    const int* src = eidx;
    const int* dst = eidx + E;

    // workspace layout (floats); N,E multiples of 4 -> float4 regions 16B aligned
    float* ws = (float*)d_ws;
    long off = 0;
    float*    h1   = ws + off; off += (long)N * 256;
    float*    acc1 = ws + off; off += (long)N * 256;   // becomes x after finx
    float*    h2   = ws + off; off += (long)N * HID;
    float*    acc2 = ws + off; off += (long)N * HID;   // becomes emb
    float*    as1  = ws + off; off += (long)N * HEADS;
    float*    ad1  = ws + off; off += (long)N * HEADS;
    unsigned* m1   = (unsigned*)(ws + off); off += (long)N * HEADS;
    float*    z1   = ws + off; off += (long)N * HEADS;
    float*    as2  = ws + off; off += (long)N;
    float*    ad2  = ws + off; off += (long)N;
    unsigned* m2   = (unsigned*)(ws + off); off += (long)N;
    float*    z2   = ws + off; off += (long)N;
    float*    e1   = ws + off; off += (long)E * HEADS;
    float*    e2   = ws + off; off += (long)E;
    float*    gsum = ws + off; off += 64;
    const long zero_total = off;                        // everything that needs zeroing
    off = (off + 7) & ~7L;                              // 32B align for v16h loads
    _Float16* W2p  = (_Float16*)(ws + off); off += 8192; // 8*4*32*16 halves = 32KB

    auto cdiv = [](long a, long b) { return (int)((a + b - 1) / b); };
    const int B = 256;

    // zero all accumulators / max buffers (encoded 0 < encode(-inf))
    k_zero<<<cdiv(zero_total, B), B, 0, stream>>>((unsigned*)ws, zero_total);

    // ---- layer 1 ----
    k_gemm1<<<cdiv((long)N * 256, B), B, 0, stream>>>(x_in, W1, h1, N);
    k_alpha1<<<cdiv((long)N * HEADS, B), B, 0, stream>>>(h1, aS1, aD1, as1, ad1, N);
    k_edge_max1<<<cdiv((long)E * HEADS, B), B, 0, stream>>>(src, dst, as1, ad1, e1, m1, E);
    k_edge_expsum1<<<cdiv((long)E * HEADS, B), B, 0, stream>>>(dst, e1, m1, z1, E);
    k_edge_msg1<<<cdiv((long)E * HEADS * 16, B), B, 0, stream>>>(src, dst, e1, m1, z1, h1, acc1, E);
    k_finx<<<cdiv((long)N * 256, B), B, 0, stream>>>(acc1, b1, N);

    // ---- layer 2 (WMMA GEMM) ----
    k_packW2<<<cdiv(8 * 4 * 32 * 16, B), B, 0, stream>>>(W2, W2p);
    int mtiles = (N + 15) / 16;            // one wave per 16x64 output stripe
    k_gemm2_wmma<<<cdiv((long)mtiles, 8), B, 0, stream>>>(acc1, W2p, h2, N);
    k_alpha2<<<cdiv(N, B), B, 0, stream>>>(h2, aS2, aD2, as2, ad2, N);
    k_edge_max2<<<cdiv(E, B), B, 0, stream>>>(src, dst, as2, ad2, e2, m2, E);
    k_edge_expsum2<<<cdiv(E, B), B, 0, stream>>>(dst, e2, m2, z2, E);
    k_edge_msg2<<<cdiv((long)E * 16, B), B, 0, stream>>>(src, dst, e2, m2, z2, h2, acc2, E);
    k_finemb<<<cdiv((long)N * HID, B), B, 0, stream>>>(acc2, b2, N);

    // ---- heads ----
    k_colsum<<<256, HID, 0, stream>>>(acc2, gsum, N);
    k_heads<<<1, HID, 0, stream>>>(gsum, acc2, Wv, bv, Wp, bp, cur, out, N);
}
